// DAB_5712306504111
// MI455X (gfx1250) — matrix operand order repaired and checked by reading, not compiled, not run on hardware
//
#include <hip/hip_runtime.h>
#include <hip/hip_bf16.h>

// ---------------------------------------------------------------------------
// B=8, C=64, H=W=192. Chain: dw1 -> mix1 -> conv1 -> conv2 -> dw2 -> mix2 ->
// conv3. All GEMMs on V_WMMA_F32_16X16X32_BF16; all inter-stage activations
// stored bf16 (halves HBM traffic on a bandwidth-bound chain); conv weights /
// channel-mix matrices pre-packed to bf16 once and staged into LDS with
// GLOBAL_LOAD_ASYNC_TO_LDS_B128 (ASYNCcnt / s_wait_asynccnt path).
// ---------------------------------------------------------------------------

typedef __bf16 bf16;
typedef bf16 v16bf __attribute__((ext_vector_type(16)));
typedef bf16 v8bf  __attribute__((ext_vector_type(8)));
typedef float v8f  __attribute__((ext_vector_type(8)));

#define BN 8
#define CN 64
#define HN 192
#define WN 192
#define HWN 36864
#define KK 576   // C * 3 * 3

__device__ __forceinline__ bf16 f2bf(float f) {
    union { float f; unsigned u; } in; in.f = f;
    unsigned r = (in.u + 0x7FFFu + ((in.u >> 16) & 1u)) >> 16;
    union { unsigned short s; bf16 b; } out; out.s = (unsigned short)r;
    return out.b;
}
__device__ __forceinline__ float bf2f(bf16 b) {
    union { bf16 b; unsigned short s; } in; in.b = b;
    union { unsigned u; float f; } out; out.u = ((unsigned)in.s) << 16;
    return out.f;
}
__device__ __forceinline__ float tofloat(float x) { return x; }
__device__ __forceinline__ float tofloat(bf16 x) { return bf2f(x); }
__device__ __forceinline__ void storeval(float v, float* p) { *p = v; }
__device__ __forceinline__ void storeval(float v, bf16* p) { *p = f2bf(v); }
__device__ __forceinline__ float leakyf(float x) { return x >= 0.f ? x : 0.1f * x; }

// 32-bit LDS byte offset of a shared-memory pointer (ptrtoint of AS3 pointer).
__device__ __forceinline__ unsigned lds_off32(const void* p) {
    return (unsigned)(__SIZE_TYPE__)(const __attribute__((address_space(3))) void*)p;
}
// Async global -> LDS 16-byte copy (per active lane). Tracked by ASYNCcnt.
__device__ __forceinline__ void async_copy_b128(unsigned lds_off, const void* gptr) {
    asm volatile("global_load_async_to_lds_b128 %0, %1, off"
                 :: "v"(lds_off), "v"(gptr) : "memory");
}
__device__ __forceinline__ void wait_async0() {
    asm volatile("s_wait_asynccnt 0x0" ::: "memory");
}

// ---------------------------------------------------------------------------
// Kernel 0: pre-pack weights to WMMA K-ordering, bf16.
//   WP[conv][o*576 + (dy*3+dx)*64 + c] = w[(o*64+c)*9 + dy*3+dx]  (3 convs)
//   then cw1, cw2 (64x64 row-major) appended.
// ---------------------------------------------------------------------------
__global__ void wprep_kernel(const float* __restrict__ w1, const float* __restrict__ w2,
                             const float* __restrict__ w3, const float* __restrict__ cw1,
                             const float* __restrict__ cw2, bf16* __restrict__ WP) {
    int tid = blockIdx.x * 256 + threadIdx.x;
    if (tid < 3 * 36864) {
        int which = tid / 36864, i = tid % 36864;
        const float* w = (which == 0) ? w1 : (which == 1 ? w2 : w3);
        int o = i / KK, k = i % KK, kd = k >> 6, c = k & 63;
        WP[which * 36864 + i] = f2bf(w[((o << 6) + c) * 9 + kd]);
    } else {
        int t = tid - 3 * 36864;  // 0 .. 8191
        const float* cw = (t < 4096) ? cw1 : cw2;
        WP[3 * 36864 + t] = f2bf(cw[t & 4095]);
    }
}

// ---------------------------------------------------------------------------
// Kernel 1: per-batch dynamic parameters (tiny FC nets). One block per batch.
// ---------------------------------------------------------------------------
__global__ void params_kernel(const float* __restrict__ d,
                              const float* __restrict__ kw1a, const float* __restrict__ kw2a,
                              const float* __restrict__ ca1a, const float* __restrict__ ca2a,
                              const float* __restrict__ kw1b, const float* __restrict__ kw2b,
                              const float* __restrict__ ca1b, const float* __restrict__ ca2b,
                              const float* __restrict__ addw1, const float* __restrict__ addw2,
                              float* __restrict__ kern1, float* __restrict__ att1,
                              float* __restrict__ kern2, float* __restrict__ att2,
                              float* __restrict__ addv) {
    int b = blockIdx.x;
    __shared__ float dv[64], t1[64], t2[64], tv[64], u1[8], u2[8];
    int tid = threadIdx.x;
    if (tid < 64) dv[tid] = d[b * 64 + tid];
    __syncthreads();
    if (tid < 64) {
        float s1 = 0.f, s2 = 0.f, sv = 0.f;
        for (int j = 0; j < 64; ++j) {
            float dj = dv[j];
            s1 += dj * kw1a[tid * 64 + j];
            s2 += dj * kw1b[tid * 64 + j];
            sv += dj * addw1[tid * 64 + j];
        }
        t1[tid] = leakyf(s1); t2[tid] = leakyf(s2); tv[tid] = leakyf(sv);
    }
    if (tid < 8) {
        float sa = 0.f, sb = 0.f;
        for (int j = 0; j < 64; ++j) {
            float dj = dv[j];
            sa += dj * ca1a[tid * 64 + j];
            sb += dj * ca1b[tid * 64 + j];
        }
        u1[tid] = leakyf(sa); u2[tid] = leakyf(sb);
    }
    __syncthreads();
    for (int m = tid; m < 576; m += 256) {
        float s1 = 0.f, s2 = 0.f;
        for (int i = 0; i < 64; ++i) {
            s1 += t1[i] * kw2a[m * 64 + i];
            s2 += t2[i] * kw2b[m * 64 + i];
        }
        kern1[b * 576 + m] = s1;
        kern2[b * 576 + m] = s2;
    }
    if (tid < 64) {
        float s1 = 0.f, s2 = 0.f;
        for (int r = 0; r < 8; ++r) {
            s1 += u1[r] * ca2a[tid * 8 + r];
            s2 += u2[r] * ca2b[tid * 8 + r];
        }
        att1[b * 64 + tid] = 1.f / (1.f + __expf(-s1));
        att2[b * 64 + tid] = 1.f / (1.f + __expf(-s2));
    }
    for (int m = tid; m < 1024; m += 256) {
        float s = 0.f;
        for (int i = 0; i < 64; ++i) s += tv[i] * addw2[m * 64 + i];
        addv[b * 1024 + m] = s;
    }
}

// ---------------------------------------------------------------------------
// Kernel 2: dynamic depthwise 3x3 conv + leaky (bandwidth-bound). Input may be
// fp32 (x) or bf16 (intermediate); output bf16.
// ---------------------------------------------------------------------------
template <typename IT>
__global__ void dw_kernel(const IT* __restrict__ in, const float* __restrict__ kern,
                          bf16* __restrict__ out) {
    int plane = blockIdx.x / 144;            // b*64 + c
    int p = (blockIdx.x % 144) * 256 + threadIdx.x;
    int y = p / WN, x = p % WN;
    const float* kp = kern + plane * 9;
    const IT* ip = in + (size_t)plane * HWN;
    float acc = 0.f;
#pragma unroll
    for (int dy = 0; dy < 3; ++dy) {
        int gy = y + dy - 1;
        if ((unsigned)gy < (unsigned)HN) {
#pragma unroll
            for (int dx = 0; dx < 3; ++dx) {
                int gx = x + dx - 1;
                if ((unsigned)gx < (unsigned)WN)
                    acc += kp[dy * 3 + dx] * tofloat(ip[gy * WN + gx]);
            }
        }
    }
    out[(size_t)plane * HWN + p] = f2bf(leakyf(acc));
}

// ---------------------------------------------------------------------------
// Kernel 3: channel-mix GEMM via WMMA bf16: out = leaky(cw@y + cb + resid*att).
// cw pre-packed bf16 -> async-staged to LDS. y (bf16) transposed through VGPRs.
// ---------------------------------------------------------------------------
template <typename RT>
__global__ void mix_kernel(const bf16* __restrict__ y, const bf16* __restrict__ cwp,
                           const float* __restrict__ cb, const float* __restrict__ att,
                           const RT* __restrict__ resid, bf16* __restrict__ out) {
    extern __shared__ char smem[];
    bf16* cwl = (bf16*)smem;                  // [o][c] 64x64
    bf16* yl  = (bf16*)(smem + 64 * 64 * 2);  // [pix][c] 256x64
    int b = blockIdx.x / 144;
    int pbase = (blockIdx.x % 144) * 256;
    int tid = threadIdx.x;

    // async global->LDS copy of cw (8 KB = 512 x 16B chunks)
    unsigned cwbase = lds_off32(cwl);
    for (int i = tid; i < 512; i += 256)
        async_copy_b128(cwbase + i * 16, (const char*)cwp + i * 16);

    const bf16* yb = y + (size_t)b * CN * HWN;
    for (int i = tid; i < 256 * 64; i += 256) {
        int c = i >> 8, p = i & 255;                       // p over tid -> coalesced
        yl[p * 64 + c] = yb[(size_t)c * HWN + pbase + p];
    }
    wait_async0();
    __syncthreads();

    int wave = tid >> 5, lane = tid & 31;
    int laneHi = lane >> 4, col = lane & 15;
    int mBase = (wave >> 1) * 16;
    int ntBase = (wave & 1) * 8;

    v8f acc[8];
    v8f zero = {0.f, 0.f, 0.f, 0.f, 0.f, 0.f, 0.f, 0.f};
#pragma unroll
    for (int t = 0; t < 8; ++t) acc[t] = zero;

#pragma unroll
    for (int ks = 0; ks < 2; ++ks) {
        int kBase = ks * 32;
        const bf16* arow = cwl + (mBase + col) * 64 + kBase + laneHi * 16;
        v8bf alo = *(const v8bf*)arow;
        v8bf ahi = *(const v8bf*)(arow + 8);
        v16bf a;
#pragma unroll
        for (int i = 0; i < 8; ++i) { a[i] = alo[i]; a[i + 8] = ahi[i]; }
#pragma unroll
        for (int t = 0; t < 8; ++t) {
            int n = (ntBase + t) * 16 + col;
            v16bf bfrag = *(const v16bf*)(yl + n * 64 + kBase + laneHi * 16);
            acc[t] = __builtin_amdgcn_wmma_f32_16x16x32_bf16(
                false, a, false, bfrag, (short)0, acc[t], false, false);
        }
    }

    bf16* ob = out + (size_t)b * CN * HWN;
    const RT* rb = resid + (size_t)b * CN * HWN;
#pragma unroll
    for (int t = 0; t < 8; ++t) {
        int n = (ntBase + t) * 16 + col;
        size_t p = (size_t)pbase + n;
#pragma unroll
        for (int r = 0; r < 8; ++r) {
            int o = mBase + laneHi * 8 + r;
            float v = acc[t][r] + cb[o] + tofloat(rb[(size_t)o * HWN + p]) * att[b * 64 + o];
            ob[(size_t)o * HWN + p] = f2bf(leakyf(v));
        }
    }
}

// ---------------------------------------------------------------------------
// Kernel 4: dense 3x3 conv as implicit GEMM (M=64, K=576, N=256 px/block).
// Weights pre-packed bf16 -> async global->LDS (72 KB). Patch (bf16) staged
// through VGPRs with transpose + zero padding.
// MODE 0 = conv1 (bias, leaky, +upsampled add) -> bf16
// MODE 1 = conv2 (bias) -> bf16
// MODE 2 = conv3 (bias, +x residual) -> fp32 d_out
// ---------------------------------------------------------------------------
template <int MODE, typename OT>
__global__ void conv3_kernel(const bf16* __restrict__ in, const bf16* __restrict__ wpre,
                             const float* __restrict__ bias, const float* __restrict__ addv,
                             const float* __restrict__ resid, OT* __restrict__ out) {
    extern __shared__ char smem[];
    bf16* wl = (bf16*)smem;                    // [o][kd*64+c] 64x576 (72 KB)
    bf16* pl = (bf16*)(smem + 64 * KK * 2);    // [sy][sx][c] 18x18x64 (40.5 KB)
    int b = blockIdx.x / 144;
    int tile = blockIdx.x % 144;
    int ty = (tile / 12) * 16, tx = (tile % 12) * 16;
    int tid = threadIdx.x;

    // async global->LDS weight stream: 73728 B = 4608 x 16B chunks, 18/lane
    unsigned wbase = lds_off32(wl);
    for (int i = tid; i < 4608; i += 256)
        async_copy_b128(wbase + i * 16, (const char*)wpre + i * 16);

    // stage 18x18 halo patch (bf16, zero-padded borders, pixel->channel transpose)
    const bf16 ZB = f2bf(0.f);
    const bf16* ib = in + (size_t)b * CN * HWN;
    for (int i = tid; i < 64 * 324; i += 256) {
        int c = i / 324, sp = i % 324;
        int sy = sp / 18, sx = sp % 18;
        int gy = ty + sy - 1, gx = tx + sx - 1;
        bf16 v = ((unsigned)gy < (unsigned)HN && (unsigned)gx < (unsigned)WN)
                     ? ib[(size_t)c * HWN + gy * WN + gx] : ZB;
        pl[sp * 64 + c] = v;
    }
    wait_async0();
    __syncthreads();

    int wave = tid >> 5, lane = tid & 31;
    int laneHi = lane >> 4, col = lane & 15;
    int mBase = (wave >> 1) * 16;
    int ntBase = (wave & 1) * 8;   // ntile == output row py within the 16x16 tile

    v8f acc[8];
    v8f zero = {0.f, 0.f, 0.f, 0.f, 0.f, 0.f, 0.f, 0.f};
#pragma unroll
    for (int t = 0; t < 8; ++t) acc[t] = zero;

    for (int s = 0; s < 18; ++s) {          // 18 K-steps of 32
        int kd = s >> 1;
        int dy = kd / 3, dx = kd % 3;
        int cBase = (s & 1) * 32;
        int kOff = kd * 64 + cBase;
        const bf16* arow = wl + (mBase + col) * KK + kOff + laneHi * 16;
        v8bf alo = *(const v8bf*)arow;
        v8bf ahi = *(const v8bf*)(arow + 8);
        v16bf a;
#pragma unroll
        for (int i = 0; i < 8; ++i) { a[i] = alo[i]; a[i + 8] = ahi[i]; }
#pragma unroll
        for (int t = 0; t < 8; ++t) {
            int py = ntBase + t;
            int sy = py + dy, sx = col + dx;
            v16bf bfrag = *(const v16bf*)(pl + (sy * 18 + sx) * 64 + cBase + laneHi * 16);
            acc[t] = __builtin_amdgcn_wmma_f32_16x16x32_bf16(
                false, a, false, bfrag, (short)0, acc[t], false, false);
        }
    }

    OT* ob = out + (size_t)b * CN * HWN;
    const float* rb = (MODE == 2) ? resid + (size_t)b * CN * HWN : nullptr;
#pragma unroll
    for (int t = 0; t < 8; ++t) {
        int py = ntBase + t;
        int gy = ty + py, gx = tx + col;
#pragma unroll
        for (int r = 0; r < 8; ++r) {
            int o = mBase + laneHi * 8 + r;
            float v = acc[t][r] + bias[o];
            if (MODE == 0) v = leakyf(v) + addv[b * 1024 + (gy / 6) * 32 + (gx / 6)];
            if (MODE == 2) v += rb[(size_t)o * HWN + gy * WN + gx];
            storeval(v, &ob[(size_t)o * HWN + gy * WN + gx]);
        }
    }
}

// ---------------------------------------------------------------------------
// Launcher. Workspace (need ~114 MB):
//   floats [0, 20480)             : dynamic params
//   bf16   next 131072 elements   : pre-packed weights (3 convs + cw1 + cw2)
//   3 rotating bf16 activation buffers of 8*64*192*192 (~37.7 MB each)
// ---------------------------------------------------------------------------
extern "C" void kernel_launch(void* const* d_in, const int* in_sizes, int n_in,
                              void* d_out, int out_size, void* d_ws, size_t ws_size,
                              hipStream_t stream) {
    (void)in_sizes; (void)n_in; (void)out_size; (void)ws_size;
    const float* x        = (const float*)d_in[0];
    const float* d        = (const float*)d_in[1];
    const float* da1_kw1  = (const float*)d_in[2];
    const float* da1_kw2  = (const float*)d_in[3];
    const float* da1_cw   = (const float*)d_in[4];
    const float* da1_cb   = (const float*)d_in[5];
    const float* da1_ca1  = (const float*)d_in[6];
    const float* da1_ca2  = (const float*)d_in[7];
    const float* da2_kw1  = (const float*)d_in[8];
    const float* da2_kw2  = (const float*)d_in[9];
    const float* da2_cw   = (const float*)d_in[10];
    const float* da2_cb   = (const float*)d_in[11];
    const float* da2_ca1  = (const float*)d_in[12];
    const float* da2_ca2  = (const float*)d_in[13];
    const float* conv1_w  = (const float*)d_in[14];
    const float* conv1_b  = (const float*)d_in[15];
    const float* conv2_w  = (const float*)d_in[16];
    const float* conv2_b  = (const float*)d_in[17];
    const float* conv3_w  = (const float*)d_in[18];
    const float* conv3_b  = (const float*)d_in[19];
    const float* add_w1   = (const float*)d_in[20];
    const float* add_w2   = (const float*)d_in[21];

    float* P     = (float*)d_ws;
    float* kern1 = P;                 // 8*576
    float* att1  = P + 4608;          // 8*64
    float* kern2 = P + 5120;          // 8*576
    float* att2  = P + 9728;          // 8*64
    float* addv  = P + 10240;         // 8*1024; params end at 18432
    bf16* WP     = (bf16*)(P + 20480);
    bf16* wp1    = WP;                // 64*576
    bf16* wp2    = WP + 36864;
    bf16* wp3    = WP + 73728;
    bf16* cw1p   = WP + 110592;       // 64*64
    bf16* cw2p   = WP + 114688;
    const size_t NT = (size_t)BN * CN * HWN;
    bf16* buf0 = WP + 131072;
    bf16* buf1 = buf0 + NT;
    bf16* buf2 = buf1 + NT;

    const int GEMM_BLOCKS = BN * 144;           // 1152
    const int DW_BLOCKS   = BN * CN * 144;      // 73728
    const size_t MIX_LDS  = (64 * 64 + 256 * 64) * sizeof(bf16);   // 40 KB
    const size_t CONV_LDS = (64 * KK + 324 * 64) * sizeof(bf16);   // 112.5 KB

    // Pre-pack weights (bf16, WMMA K-order) + dynamic params.
    wprep_kernel<<<464, 256, 0, stream>>>(conv1_w, conv2_w, conv3_w, da1_cw, da2_cw, WP);
    params_kernel<<<BN, 256, 0, stream>>>(d, da1_kw1, da1_kw2, da1_ca1, da1_ca2,
                                          da2_kw1, da2_kw2, da2_ca1, da2_ca2,
                                          add_w1, add_w2,
                                          kern1, att1, kern2, att2, addv);
    // da1: depthwise(x fp32)+leaky -> buf0; mix + x*att + leaky -> buf1
    dw_kernel<float><<<DW_BLOCKS, 256, 0, stream>>>(x, kern1, buf0);
    mix_kernel<float><<<GEMM_BLOCKS, 256, MIX_LDS, stream>>>(buf0, cw1p, da1_cb, att1, x, buf1);
    // conv1: leaky + upsampled add -> buf0
    conv3_kernel<0, bf16><<<GEMM_BLOCKS, 256, CONV_LDS, stream>>>(buf1, wp1, conv1_b, addv,
                                                                  nullptr, buf0);
    // conv2: plain -> buf1
    conv3_kernel<1, bf16><<<GEMM_BLOCKS, 256, CONV_LDS, stream>>>(buf0, wp2, conv2_b, nullptr,
                                                                  nullptr, buf1);
    // da2: depthwise(buf1)+leaky -> buf0; mix + buf1*att + leaky -> buf2
    dw_kernel<bf16><<<DW_BLOCKS, 256, 0, stream>>>(buf1, kern2, buf0);
    mix_kernel<bf16><<<GEMM_BLOCKS, 256, MIX_LDS, stream>>>(buf0, cw2p, da2_cb, att2, buf1, buf2);
    // conv3: + x residual -> fp32 d_out
    conv3_kernel<2, float><<<GEMM_BLOCKS, 256, CONV_LDS, stream>>>(buf2, wp3, conv3_b, nullptr,
                                                                   x, (float*)d_out);
}
